// GEXTEmbeddings_15599321219250
// MI455X (gfx1250) — matrix operand order repaired and checked by reading, not compiled, not run on hardware
//
#include <hip/hip_runtime.h>
#include <hip/hip_bf16.h>

#define Bq 8
#define Gq 20000
#define Hq 512
#define Kq 256
#define NCHUNK (Gq / 32)

typedef __attribute__((ext_vector_type(16))) __bf16 v16bf;
typedef __attribute__((ext_vector_type(8)))  __bf16 v8bf;
typedef __attribute__((ext_vector_type(8)))  float  v8f;

// Fold mask: sge = ge*(1-m); sid = (masked id) * Hq  (pre-scaled row offset)
__global__ void prep_kernel(const float* __restrict__ gex, const int* __restrict__ ids,
                            const int* __restrict__ mask, const int* __restrict__ mid_p,
                            float* __restrict__ sge, unsigned* __restrict__ sid) {
  int i = blockIdx.x * 256 + threadIdx.x;
  if (i < Bq * Gq) {
    int m = mask[i];
    sge[i] = m ? 0.0f : gex[i];
    sid[i] = (unsigned)(m ? mid_p[0] : ids[i]) * (unsigned)Hq;
  }
}

// recip[b*K+k] = 1 / sum_g group_mtx[b,k,g]
__global__ void nsum_kernel(const float* __restrict__ grp, float* __restrict__ recip) {
  __shared__ float red[256];
  const float* row = grp + (size_t)blockIdx.x * Gq;
  float s = 0.f;
  for (int g = threadIdx.x; g < Gq; g += 256) s += row[g];
  red[threadIdx.x] = s; __syncthreads();
  for (int off = 128; off; off >>= 1) {
    if (threadIdx.x < off) red[threadIdx.x] += red[threadIdx.x + off];
    __syncthreads();
  }
  if (!threadIdx.x) recip[blockIdx.x] = 1.0f / red[0];
}

// out[b,0,:] = cls_token
__global__ void cls_kernel(const float* __restrict__ cls_token, float* __restrict__ out) {
  int i = blockIdx.x * 256 + threadIdx.x;       // B*H = 4096
  int b = i >> 9, h = i & 511;
  out[((size_t)b * (Kq + 1)) * Hq + h] = cls_token[h];
}

// Fused gather + scale + batched GEMM via bf16 WMMA, double-buffered LDS.
// Grid: (H/128, K/64, B). Block: 256 threads = 8 waves as 2(M) x 4(N),
// each wave: 32x32 output tile = 2x2 v_wmma_f32_16x16x32_bf16 accumulators.
__global__ __launch_bounds__(256) void gemm_kernel(
    const float* __restrict__ grp, const float* __restrict__ emb,
    const float* __restrict__ sge, const unsigned* __restrict__ sid,
    const float* __restrict__ recip, float* __restrict__ out) {
  // fragments pre-swizzled in WMMA lane layout: [buf][tile][lane][16 bf16]
  __shared__ __align__(32) unsigned short Af[2][4][32][16];   // 8 KB
  __shared__ __align__(32) unsigned short Bf[2][8][32][16];   // 16 KB

  const int b  = blockIdx.z;
  const int k0 = blockIdx.y * 64;
  const int n0 = blockIdx.x * 128;
  const int t  = threadIdx.x;
  const int lane  = t & 31;
  const int w     = t >> 5;
  const int waveM = w >> 2;          // 0..1
  const int waveN = w & 3;           // 0..3

  // --- A staging map: thread -> (row m, col group c), one 16B store to Af
  const int am = t >> 2;                         // 0..63
  const int ac = (t & 3) << 3;                   // 0,8,16,24
  const int aLane = (am & 15) + ((ac & 16) ? 16 : 0);
  const int aE0   = (ac & 8) ? 8 : 0;
  const int aMt   = am >> 4;
  const float* aBase = grp + ((size_t)b * Kq + (k0 + am)) * Gq + ac;

  // --- B staging map: thread -> column n, 16 K-rows; one 32B store to Bf
  const int bn   = t & 127;                      // 0..127 (column within tile)
  const int kgrp = t >> 7;                       // 0..1 (wave-uniform)
  const int rsel = __builtin_amdgcn_readfirstlane(kgrp); // force scalar path
  const int bLane = (bn & 15) + kgrp * 16;
  const int bNt   = bn >> 4;
  const int sgBase = b * Gq;
  const unsigned nOff = (unsigned)(n0 + bn);     // per-thread constant column

  v8f acc00 = {}, acc01 = {}, acc10 = {}, acc11 = {};

  auto stage = [&](int g0, int buf) {
    // A tile 64x32 (f32 -> bf16, fragment layout)
    {
      const float4 f0 = *(const float4*)(aBase + g0);
      const float4 f1 = *(const float4*)(aBase + g0 + 4);
      v8bf ap;
      ap[0] = (__bf16)f0.x; ap[1] = (__bf16)f0.y;
      ap[2] = (__bf16)f0.z; ap[3] = (__bf16)f0.w;
      ap[4] = (__bf16)f1.x; ap[5] = (__bf16)f1.y;
      ap[6] = (__bf16)f1.z; ap[7] = (__bf16)f1.w;
      *(v8bf*)&Af[buf][aMt][aLane][aE0] = ap;
    }
    // B tile 32x128: gather emb rows (u32 offsets), scale by sge, bf16
    {
      const int sg0 = sgBase + g0 + rsel * 16;   // wave-uniform base
      v16bf bp;
      #pragma unroll
      for (int i = 0; i < 16; ++i) {
        float    ge  = sge[sg0 + i];
        unsigned off = sid[sg0 + i] + nOff;      // pre-scaled row offset + col
        bp[i] = (__bf16)(ge * emb[off]);
      }
      *(v16bf*)&Bf[buf][bNt][bLane][0] = bp;
    }
  };

  stage(0, 0);
  for (int c = 0; c < NCHUNK; ++c) {
    __syncthreads();
    const int buf = c & 1;
    v16bf a0 = *(const v16bf*)Af[buf][2 * waveM    ][lane];
    v16bf a1 = *(const v16bf*)Af[buf][2 * waveM + 1][lane];
    v16bf b0 = *(const v16bf*)Bf[buf][2 * waveN    ][lane];
    v16bf b1 = *(const v16bf*)Bf[buf][2 * waveN + 1][lane];
    if (c + 1 < NCHUNK) stage((c + 1) * 32, buf ^ 1);
    acc00 = __builtin_amdgcn_wmma_f32_16x16x32_bf16(false, a0, false, b0, (short)0, acc00, false, false);
    acc01 = __builtin_amdgcn_wmma_f32_16x16x32_bf16(false, a0, false, b1, (short)0, acc01, false, false);
    acc10 = __builtin_amdgcn_wmma_f32_16x16x32_bf16(false, a1, false, b0, (short)0, acc10, false, false);
    acc11 = __builtin_amdgcn_wmma_f32_16x16x32_bf16(false, a1, false, b1, (short)0, acc11, false, false);
  }

  // epilogue: C layout — VGPR v: M = v (+8 for lanes 16..31), N = lane%16
  const int colLo = lane & 15;
  const int rOff  = (lane < 16) ? 0 : 8;
  #pragma unroll
  for (int i = 0; i < 2; ++i) {
    const int rowBase = k0 + (2 * waveM + i) * 16 + rOff;
    #pragma unroll
    for (int j = 0; j < 2; ++j) {
      const int col = n0 + (2 * waveN + j) * 16 + colLo;
      v8f a = (i == 0) ? (j == 0 ? acc00 : acc01) : (j == 0 ? acc10 : acc11);
      #pragma unroll
      for (int v8 = 0; v8 < 8; ++v8) {
        int kRow = rowBase + v8;
        float sc = recip[b * Kq + kRow];
        out[((size_t)b * (Kq + 1) + 1 + kRow) * Hq + col] = a[v8] * sc;
      }
    }
  }
}

extern "C" void kernel_launch(void* const* d_in, const int* in_sizes, int n_in,
                              void* d_out, int out_size, void* d_ws, size_t ws_size,
                              hipStream_t stream) {
  const float* gex  = (const float*)d_in[0];
  const int*   ids  = (const int*)d_in[1];
  const int*   mask = (const int*)d_in[2];
  const float* grp  = (const float*)d_in[3];
  const float* emb  = (const float*)d_in[4];
  const float* cls  = (const float*)d_in[5];
  const int*   mid  = (const int*)d_in[6];
  float* out = (float*)d_out;

  float*    sge   = (float*)d_ws;                                    // B*G f32
  unsigned* sid   = (unsigned*)((char*)d_ws + (size_t)Bq * Gq * 4);  // B*G u32
  float*    recip = (float*)((char*)d_ws + (size_t)2 * Bq * Gq * 4); // B*K f32

  prep_kernel<<<(Bq * Gq + 255) / 256, 256, 0, stream>>>(gex, ids, mask, mid, sge, sid);
  nsum_kernel<<<Bq * Kq, 256, 0, stream>>>(grp, recip);
  cls_kernel<<<(Bq * Hq) / 256, 256, 0, stream>>>(cls, out);
  gemm_kernel<<<dim3(Hq / 128, Kq / 64, Bq), 256, 0, stream>>>(grp, emb, sge, sid, recip, out);
}
